// SAG_88081189307161
// MI455X (gfx1250) — matrix operand order repaired and checked by reading, not compiled, not run on hardware
//
#include <hip/hip_runtime.h>

// ---------------------------------------------------------------------------
// CDNA5 (gfx1250) implementation of the change-detection block.
// GEMM-shaped work (1x1 convs, implicit-GEMM 3x3/5x5 convs, attention einsums)
// uses v_wmma_f32_16x16x32_f16 with LDS-staged f16 tiles:
//   global_load_b128 (f32, coalesced) -> v_cvt_pk_f16_f32 -> ds_store
//   ds_load_b128 fragment reads -> v_wmma
// Epilogue variants are compile-time template bits (straight-line code).
// Shapes: B=4, C_IN=256, C=128, H=W=64, HW=4096, N=1024 (pooled).
// ---------------------------------------------------------------------------

typedef __attribute__((ext_vector_type(16))) _Float16 v16h;
typedef __attribute__((ext_vector_type(8)))  _Float16 v8h;
typedef __attribute__((ext_vector_type(2)))  _Float16 v2h;
typedef __attribute__((ext_vector_type(8)))  float    v8f;

__device__ inline v8f wmma_f16(v16h a, v16h b, v8f c) {
  // 8 args: (neg_a, A, neg_b, B, c_mod, C, reuse_a, reuse_b)
  return __builtin_amdgcn_wmma_f32_16x16x32_f16(false, a, false, b,
                                                (short)0, c, false, false);
}

__device__ inline v16h cat8(v8h a, v8h b) {
  return __builtin_shufflevector(a, b, 0, 1, 2, 3, 4, 5, 6, 7,
                                 8, 9, 10, 11, 12, 13, 14, 15);
}

// epilogue bits
constexpr int EP_BIAS  = 1;
constexpr int EP_RELU  = 2;
constexpr int EP_SCALE = 4;
constexpr int EP_RESID = 8;

// ---------------------------------------------------------------------------
// Batched WMMA GEMM, N fixed at 4096, A-strides + epilogue as template consts.
//   OUT[b][m][n] = epi( sum_k A[b][m*RSA + k*KSA] * B[b][k*4096 + n] )
// epilogue: +bias[m]; relu; *scale[m]+offs[m]; +resid[b][m*4096+n]
// block = 128 threads (4 waves, 2x2), WG tile 64x64, wave tile 32x32.
// Requires M % 64 == 0, K % 32 == 0.
// ---------------------------------------------------------------------------
template <int RSA, int KSA, int EPI>
__global__ __launch_bounds__(128) void gemm_wmma(
    const float* __restrict__ A, long bsA,
    const float* __restrict__ Bm, long bsB,
    float* __restrict__ Out, long bsO,
    int M, int K,
    const float* __restrict__ bias,
    const float* __restrict__ scale, const float* __restrict__ offs,
    const float* __restrict__ resid, long bsR)
{
  constexpr int N  = 4096;
  constexpr int PA = 40;   // f16 pitch (80B): 16B-aligned chunks, low conflicts
  constexpr int PB = 40;
  __shared__ __align__(16) _Float16 lA[64 * PA];  // [m][k]  (64 x 32 used)
  __shared__ __align__(16) _Float16 lB[64 * PB];  // [n][k]  transposed

  const int bz = blockIdx.z;
  const int mblk = blockIdx.y * 64;
  const int nblk = blockIdx.x * 64;
  A  += (long)bz * bsA;
  Bm += (long)bz * bsB + nblk;
  Out += (long)bz * bsO;
  if constexpr (EPI & EP_RESID) resid += (long)bz * bsR;

  const int tid  = threadIdx.x;
  const int wave = tid >> 5;
  const int lane = tid & 31;
  const int l15  = lane & 15;
  const int lhi  = lane >> 4;
  const int m0w  = (wave >> 1) * 32;   // wave tile origin in WG tile
  const int n0w  = (wave & 1) * 32;

  // staging roles
  const int wa_m = tid >> 1;           // 0..63  (A row in tile)
  const int wa_k = (tid & 1) * 16;     // 0/16   (A k-half)
  const int wb_kp = tid >> 3;          // 0..15  (B k-pair)
  const int wb_n  = (tid & 7) * 8;     // 0..56  (B n-group)

  v8f acc[2][2] = {};

  for (int k0 = 0; k0 < K; k0 += 32) {
    // ---- stage A tile (64 x 32 f16) ----
    {
      _Float16 tmp[16];
      if constexpr (KSA == 1) {
        const float4* s4 =
            (const float4*)(A + (long)(mblk + wa_m) * RSA + (k0 + wa_k));
#pragma unroll
        for (int j = 0; j < 4; ++j) {
          float4 v = s4[j];
          tmp[j * 4 + 0] = (_Float16)v.x;
          tmp[j * 4 + 1] = (_Float16)v.y;
          tmp[j * 4 + 2] = (_Float16)v.z;
          tmp[j * 4 + 3] = (_Float16)v.w;
        }
      } else {
        const float* s = A + (long)(mblk + wa_m) * RSA + (long)(k0 + wa_k) * KSA;
#pragma unroll
        for (int j = 0; j < 16; ++j) tmp[j] = (_Float16)s[(long)j * KSA];
      }
      v8h* dst = (v8h*)&lA[wa_m * PA + wa_k];
      v8h p0, p1;
#pragma unroll
      for (int j = 0; j < 8; ++j) { p0[j] = tmp[j]; p1[j] = tmp[j + 8]; }
      dst[0] = p0;
      dst[1] = p1;
    }
    // ---- stage B tile transposed (64 n x 32 k f16) ----
    {
      const float4* s0 = (const float4*)(Bm + (long)(k0 + 2 * wb_kp) * N + wb_n);
      const float4* s1 = (const float4*)(Bm + (long)(k0 + 2 * wb_kp + 1) * N + wb_n);
      float4 r0a = s0[0], r0b = s0[1];
      float4 r1a = s1[0], r1b = s1[1];
      float k0v[8] = {r0a.x, r0a.y, r0a.z, r0a.w, r0b.x, r0b.y, r0b.z, r0b.w};
      float k1v[8] = {r1a.x, r1a.y, r1a.z, r1a.w, r1b.x, r1b.y, r1b.z, r1b.w};
#pragma unroll
      for (int j = 0; j < 8; ++j) {
        v2h p;
        p[0] = (_Float16)k0v[j];
        p[1] = (_Float16)k1v[j];
        *(v2h*)&lB[(wb_n + j) * PB + 2 * wb_kp] = p;
      }
    }
    __syncthreads();

    // ---- fragment reads (ds_load_b128) + WMMA ----
    v16h afr[2], bfr[2];
#pragma unroll
    for (int t = 0; t < 2; ++t) {
      const _Float16* ap = &lA[(m0w + t * 16 + l15) * PA];
      v8h lo = *(const v8h*)(ap + lhi * 8);        // K = lhi*8 .. +7
      v8h hi = *(const v8h*)(ap + 16 + lhi * 8);   // K = 16+lhi*8 .. +7
      afr[t] = cat8(lo, hi);
    }
#pragma unroll
    for (int t = 0; t < 2; ++t) {
      const _Float16* bp = &lB[(n0w + t * 16 + l15) * PB + lhi * 8];
      v8h lo = *(const v8h*)bp;                    // K = lhi*8 .. +7
      v8h hi = *(const v8h*)(bp + 16);             // K = 16+lhi*8 .. +7
      bfr[t] = cat8(lo, hi);
    }
#pragma unroll
    for (int i = 0; i < 2; ++i)
#pragma unroll
      for (int j = 0; j < 2; ++j)
        acc[i][j] = wmma_f16(afr[i], bfr[j], acc[i][j]);
    __syncthreads();
  }

  // ---- straight-line epilogue ----
  // C/D layout: lanes 0-15: N=lane, VGPR r -> M=r; lanes 16-31: M=8+r.
#pragma unroll
  for (int i = 0; i < 2; ++i) {
    const int mbase = mblk + m0w + i * 16 + lhi * 8;
    float bb[8], ss[8], oo[8];
    if constexpr (EPI & EP_BIAS) {
      *(float4*)&bb[0] = *(const float4*)(bias + mbase);
      *(float4*)&bb[4] = *(const float4*)(bias + mbase + 4);
    }
    if constexpr (EPI & EP_SCALE) {
      *(float4*)&ss[0] = *(const float4*)(scale + mbase);
      *(float4*)&ss[4] = *(const float4*)(scale + mbase + 4);
      *(float4*)&oo[0] = *(const float4*)(offs + mbase);
      *(float4*)&oo[4] = *(const float4*)(offs + mbase + 4);
    }
#pragma unroll
    for (int j = 0; j < 2; ++j) {
      const int n = nblk + n0w + j * 16 + l15;
      float* obase = Out + (long)mbase * N + n;
      const float* rbase =
          (EPI & EP_RESID) ? (resid + (long)mbase * N + n) : nullptr;
#pragma unroll
      for (int r = 0; r < 8; ++r) {
        float v = acc[i][j][r];
        if constexpr (EPI & EP_BIAS)  v += bb[r];
        if constexpr (EPI & EP_RELU)  v = fmaxf(v, 0.f);
        if constexpr (EPI & EP_SCALE) v = v * ss[r] + oo[r];
        if constexpr (EPI & EP_RESID) v += rbase[r * N];
        obase[r * N] = v;
      }
    }
  }
}

// ---------------------------------------------------------------------------
// Implicit-GEMM "same" convolution KSxKS, Cin=128, H=W=64.
// One wave per block; wave computes 16 out-channels x 16 spatial (one row h,
// 16 consecutive w). Loops taps(KS*KS) x K-blocks(32 in-channels) of WMMA.
// Boundary handled by clamped address + cndmask (no exec-masked loads).
// STDC=true applies relu + per-channel affine (eval-mode BN).
// grid: x = (W/16)*H = 256 spatial tiles, y = Cout/16, z = batch.
// ---------------------------------------------------------------------------
template <int KS, bool STDC>
__global__ __launch_bounds__(32) void conv_wmma(
    const float* __restrict__ In, long bsIn,
    const float* __restrict__ Wt,          // [Cout][128][KS][KS]
    float* __restrict__ Out, long bsOut,
    const float* __restrict__ bias,
    const float* __restrict__ scale, const float* __restrict__ offs)
{
  constexpr int Cin = 128, Wd = 64, Hd = 64, HW = 4096, P = KS / 2, KK = KS * KS;
  const float* in = In + (long)blockIdx.z * bsIn;
  float* out = Out + (long)blockIdx.z * bsOut;

  const int h   = blockIdx.x >> 2;
  const int w0  = (blockIdx.x & 3) * 16;
  const int co0 = blockIdx.y * 16;
  const int lane = threadIdx.x & 31;
  const int l15  = lane & 15;
  const int lhi  = lane >> 4;

  v8f acc = {};
#pragma unroll
  for (int dy = 0; dy < KS; ++dy) {
    const int hy = h + dy - P;
    const bool rowok = (hy >= 0) && (hy < Hd);
    const int hyc = min(max(hy, 0), Hd - 1);
#pragma unroll
    for (int dx = 0; dx < KS; ++dx) {
      const int wx = w0 + l15 + dx - P;               // per-lane column
      const bool ok = rowok && (wx >= 0) && (wx < Wd);
      const int wxc = min(max(wx, 0), Wd - 1);
      const float* inbase = in + hyc * Wd + wxc;      // always-valid base
      const float* wtap = Wt + (long)(co0 + l15) * Cin * KK + dy * KS + dx;
      for (int k0 = 0; k0 < Cin; k0 += 32) {
        v16h af, bf;
        const int kb = k0 + lhi * 8;
        const float* wp = wtap + (long)kb * KK;
#pragma unroll
        for (int i = 0; i < 8; ++i) {
          af[i]     = (_Float16)wp[i * KK];
          af[i + 8] = (_Float16)wp[(16 + i) * KK];
        }
        const float* ip = inbase + (long)(k0 + lhi * 8) * HW;
#pragma unroll
        for (int i = 0; i < 16; ++i) {
          float v = ip[(i < 8 ? i : i + 8) * HW];     // imm-offset load
          bf[i] = (_Float16)(ok ? v : 0.f);           // cndmask, uniform exec
        }
        acc = wmma_f16(af, bf, acc);
      }
    }
  }
  const int mbase = co0 + lhi * 8;
  float bb[8], ss[8], oo[8];
  *(float4*)&bb[0] = *(const float4*)(bias + mbase);
  *(float4*)&bb[4] = *(const float4*)(bias + mbase + 4);
  if constexpr (STDC) {
    *(float4*)&ss[0] = *(const float4*)(scale + mbase);
    *(float4*)&ss[4] = *(const float4*)(scale + mbase + 4);
    *(float4*)&oo[0] = *(const float4*)(offs + mbase);
    *(float4*)&oo[4] = *(const float4*)(offs + mbase + 4);
  }
  float* obase = out + (long)mbase * HW + h * Wd + w0 + l15;
#pragma unroll
  for (int r = 0; r < 8; ++r) {
    float v = acc[r] + bb[r];
    if constexpr (STDC) v = fmaxf(v, 0.f) * ss[r] + oo[r];
    obase[r * HW] = v;
  }
}

// ---------------------------------------------------------------------------
// Cosine similarity over groups of 128 consecutive floats (raw-reshape view).
// One wave per (b,n) group.
// ---------------------------------------------------------------------------
__global__ __launch_bounds__(256) void cosine_sim(
    const float* __restrict__ r1, const float* __restrict__ r2,
    float* __restrict__ sim)
{
  const long gw = (long)(blockIdx.x * blockDim.x + threadIdx.x) >> 5;
  const int lane = threadIdx.x & 31;
  if (gw >= 4L * 4096) return;
  const float* p1 = r1 + gw * 128;
  const float* p2 = r2 + gw * 128;
  float dot = 0.f, n1 = 0.f, n2 = 0.f;
#pragma unroll
  for (int i = lane; i < 128; i += 32) {
    float a = p1[i], b = p2[i];
    dot += a * b; n1 += a * a; n2 += b * b;
  }
#pragma unroll
  for (int s = 16; s; s >>= 1) {
    dot += __shfl_xor(dot, s, 32);
    n1  += __shfl_xor(n1,  s, 32);
    n2  += __shfl_xor(n2,  s, 32);
  }
  if (lane == 0)
    sim[gw] = dot / fmaxf(sqrtf(n1) * sqrtf(n2), 1e-8f);
}

// ---------------------------------------------------------------------------
// dif1/dif2 (concatenated [B,256,HW]) from gated reshape + residual.
// ---------------------------------------------------------------------------
__global__ __launch_bounds__(256) void make_difcat(
    const float* __restrict__ r1, const float* __restrict__ r2,
    const float* __restrict__ sim, float* __restrict__ difcat)
{
  const long idx = (long)blockIdx.x * blockDim.x + threadIdx.x;  // B*128*4096
  if (idx >= 4L * 128 * 4096) return;
  const long b = idx >> 19;
  const long rem = idx & 524287;
  const int c = (int)(rem >> 12);
  const int n = (int)(rem & 4095);
  const float d = 1.f - sim[b * 4096 + n];
  const long fb = b * 524288;
  const float g1 = r1[fb + (long)n * 128 + c];   // f1[b,n,c] (raw reshape)
  const float g2 = r2[fb + (long)n * 128 + c];
  const float rr1 = r1[idx];                      // res[b,c,n]
  const float rr2 = r2[idx];
  const long ob = b * (2L * 524288);
  difcat[ob + (long)c * 4096 + n]         = g1 * d + rr1;
  difcat[ob + (long)(c + 128) * 4096 + n] = g2 * d + rr2;
}

// ---------------------------------------------------------------------------
// simp = avgpool2x2(|sim1 - sim2|)  -> [B,128,32,32]
// ---------------------------------------------------------------------------
__global__ __launch_bounds__(256) void make_simp(
    const float* __restrict__ r1, const float* __restrict__ r2,
    const float* __restrict__ sim, float* __restrict__ simp)
{
  const long idx = (long)blockIdx.x * blockDim.x + threadIdx.x;  // B*128*1024
  if (idx >= 4L * 128 * 1024) return;
  const int pw = (int)(idx & 31);
  const int ph = (int)((idx >> 5) & 31);
  const int c  = (int)((idx >> 10) & 127);
  const long b = idx >> 17;
  const long fb = b * 524288;
  float acc = 0.f;
#pragma unroll
  for (int dy = 0; dy < 2; ++dy)
#pragma unroll
    for (int dx = 0; dx < 2; ++dx) {
      const int n = (ph * 2 + dy) * 64 + (pw * 2 + dx);
      const float s = sim[b * 4096 + n];
      const float g1 = r1[fb + (long)n * 128 + c];
      const float g2 = r2[fb + (long)n * 128 + c];
      const float s1 = g1 * s + r1[fb + (long)c * 4096 + n];
      const float s2 = g2 * s + r2[fb + (long)c * 4096 + n];
      acc += fabsf(s1 - s2);
    }
  simp[idx] = acc * 0.25f;
}

// ---------------------------------------------------------------------------
// Row softmax over 4096 columns; one 256-thread block per row.
// ---------------------------------------------------------------------------
__global__ __launch_bounds__(256) void softmax_rows(float* __restrict__ S)
{
  const int cols = 4096;
  float* p = S + (long)blockIdx.x * cols;
  __shared__ float red[8];
  const int tid = threadIdx.x, lane = tid & 31, wv = tid >> 5;

  float mx = -3.4e38f;
  for (int i = tid; i < cols; i += 256) mx = fmaxf(mx, p[i]);
#pragma unroll
  for (int s = 16; s; s >>= 1) mx = fmaxf(mx, __shfl_xor(mx, s, 32));
  if (lane == 0) red[wv] = mx;
  __syncthreads();
  mx = red[0];
#pragma unroll
  for (int i = 1; i < 8; ++i) mx = fmaxf(mx, red[i]);
  __syncthreads();

  float sum = 0.f;
  for (int i = tid; i < cols; i += 256) {
    float e = __expf(p[i] - mx);
    p[i] = e;
    sum += e;
  }
#pragma unroll
  for (int s = 16; s; s >>= 1) sum += __shfl_xor(sum, s, 32);
  if (lane == 0) red[wv] = sum;
  __syncthreads();
  float tot = 0.f;
#pragma unroll
  for (int i = 0; i < 8; ++i) tot += red[i];
  const float inv = 1.f / tot;
  for (int i = tid; i < cols; i += 256) p[i] *= inv;
}

// ---------------------------------------------------------------------------
// Bilinear 2x upsample 64x64 -> 128x128 (half-pixel / align_corners=False).
// ---------------------------------------------------------------------------
__global__ __launch_bounds__(256) void upsample_bilinear(
    const float* __restrict__ in, float* __restrict__ out)
{
  const long idx = (long)blockIdx.x * blockDim.x + threadIdx.x;  // 4*128*128*128
  if (idx >= 4L * 128 * 128 * 128) return;
  const int x = (int)(idx & 127);
  const int y = (int)((idx >> 7) & 127);
  const long cidx = idx >> 14;                   // b*128 + c
  const float sy = (y + 0.5f) * 0.5f - 0.5f;
  const float sx = (x + 0.5f) * 0.5f - 0.5f;
  const int y0 = (int)floorf(sy);
  const int x0 = (int)floorf(sx);
  const float fy = sy - (float)y0;
  const float fx = sx - (float)x0;
  const int y0c = min(max(y0, 0), 63), y1c = min(max(y0 + 1, 0), 63);
  const int x0c = min(max(x0, 0), 63), x1c = min(max(x0 + 1, 0), 63);
  const float* pp = in + cidx * 4096;
  const float v00 = pp[y0c * 64 + x0c], v01 = pp[y0c * 64 + x1c];
  const float v10 = pp[y1c * 64 + x0c], v11 = pp[y1c * 64 + x1c];
  out[idx] = (1.f - fy) * ((1.f - fx) * v00 + fx * v01) +
             fy * ((1.f - fx) * v10 + fx * v11);
}

// ---------------------------------------------------------------------------
extern "C" void kernel_launch(void* const* d_in, const int* in_sizes, int n_in,
                              void* d_out, int out_size, void* d_ws, size_t ws_size,
                              hipStream_t stream) {
  (void)in_sizes; (void)n_in; (void)out_size; (void)ws_size;
  const float* t1       = (const float*)d_in[0];
  const float* t2       = (const float*)d_in[1];
  const float* t1_w     = (const float*)d_in[2];
  const float* t1_b     = (const float*)d_in[3];
  const float* t2_w     = (const float*)d_in[4];
  const float* t2_b     = (const float*)d_in[5];
  const float* df_res_w = (const float*)d_in[6];
  const float* df_res_b = (const float*)d_in[7];
  const float* df_res_s = (const float*)d_in[8];
  const float* df_res_o = (const float*)d_in[9];
  const float* df_b0_w  = (const float*)d_in[10];
  const float* df_b0_b  = (const float*)d_in[11];
  const float* df_b0_s  = (const float*)d_in[12];
  const float* df_b0_o  = (const float*)d_in[13];
  const float* df_b1_w  = (const float*)d_in[14];
  const float* df_b1_b  = (const float*)d_in[15];
  const float* df_b1_s  = (const float*)d_in[16];
  const float* df_b1_o  = (const float*)d_in[17];
  const float* df_fu_w  = (const float*)d_in[18];
  const float* df_fu_b  = (const float*)d_in[19];
  const float* df_fu_s  = (const float*)d_in[20];
  const float* df_fu_o  = (const float*)d_in[21];
  const float* br1_w    = (const float*)d_in[22];
  const float* br1_b    = (const float*)d_in[23];
  const float* br2_w    = (const float*)d_in[24];
  const float* br2_b    = (const float*)d_in[25];
  const float* fu_w     = (const float*)d_in[26];
  const float* fu_b     = (const float*)d_in[27];
  const float* fu_s     = (const float*)d_in[28];
  const float* fu_o     = (const float*)d_in[29];

  // Workspace layout (floats).
  float* ws     = (float*)d_ws;
  float* res1   = ws;                       // 4*128*4096
  float* res2   = res1   + 2097152;
  float* simv   = res2   + 2097152;         // 4*4096
  float* difcat = simv   + 16384;           // 4*256*4096
  float* rbuf   = difcat + 4194304;         // 4*128*4096
  float* b01    = rbuf   + 2097152;         // 4*128*4096 (b0 | b1)
  float* difbuf = b01    + 2097152;         // 4*128*4096
  float* simp   = difbuf + 2097152;         // 4*128*1024
  float* inc1   = simp   + 524288;          // 4*128*4096
  float* inc2   = inc1   + 2097152;         // 4*128*4096
  float* acat   = inc2   + 2097152;         // 4*256*4096 (a1 | a2)
  float* fusedb = acat   + 4194304;         // 4*128*4096
  float* scores = fusedb + 2097152;         // 4*1024*4096 (reused per branch)

  const dim3 gB(128);
  constexpr int EP_STD = EP_BIAS | EP_RELU | EP_SCALE;

  // res1/res2 = 1x1 conv (GEMM 128x4096x256) + bias
  gemm_wmma<256, 1, EP_BIAS><<<dim3(64, 2, 4), gB, 0, stream>>>(
      t1_w, 0, t1, 256L * 4096, res1, 128L * 4096,
      128, 256, t1_b, nullptr, nullptr, nullptr, 0);
  gemm_wmma<256, 1, EP_BIAS><<<dim3(64, 2, 4), gB, 0, stream>>>(
      t2_w, 0, t2, 256L * 4096, res2, 128L * 4096,
      128, 256, t2_b, nullptr, nullptr, nullptr, 0);

  // cosine sim over raw-reshape groups of 128
  cosine_sim<<<2048, 256, 0, stream>>>(res1, res2, simv);

  // gated difference concat + pooled |sim1-sim2|
  make_difcat<<<8192, 256, 0, stream>>>(res1, res2, simv, difcat);
  make_simp<<<2048, 256, 0, stream>>>(res1, res2, simv, simp);

  // r = stdconv1x1(difcat): GEMM 128x4096x256 + relu + affine
  gemm_wmma<256, 1, EP_STD><<<dim3(64, 2, 4), gB, 0, stream>>>(
      df_res_w, 0, difcat, 2L * 524288, rbuf, 524288,
      128, 256, df_res_b, df_res_s, df_res_o, nullptr, 0);

  // b0 = stdconv3x3(r) (64 ch), b1 = stdconv5x5(r) (64 ch) -> b01
  conv_wmma<3, true><<<dim3(256, 4, 4), 32, 0, stream>>>(
      rbuf, 524288, df_b0_w, b01, 524288, df_b0_b, df_b0_s, df_b0_o);
  conv_wmma<5, true><<<dim3(256, 4, 4), 32, 0, stream>>>(
      rbuf, 524288, df_b1_w, b01 + 64L * 4096, 524288,
      df_b1_b, df_b1_s, df_b1_o);

  // dif = stdconv1x1(b01) + r
  gemm_wmma<128, 1, EP_STD | EP_RESID><<<dim3(64, 2, 4), gB, 0, stream>>>(
      df_fu_w, 0, b01, 524288, difbuf, 524288,
      128, 128, df_fu_b, df_fu_s, df_fu_o, rbuf, 524288);

  // inc1 = conv3x3(dif)+b, inc2 = conv5x5(dif)+b (plain, 128 ch)
  conv_wmma<3, false><<<dim3(256, 8, 4), 32, 0, stream>>>(
      difbuf, 524288, br1_w, inc1, 524288, br1_b, nullptr, nullptr);
  conv_wmma<5, false><<<dim3(256, 8, 4), 32, 0, stream>>>(
      difbuf, 524288, br2_w, inc2, 524288, br2_b, nullptr, nullptr);

  // attention branches: scores = attn @ inc, softmax, out = attn^T @ a + inc
  for (int br = 0; br < 2; ++br) {
    const float* inc = br ? inc2 : inc1;
    // scores[b] (1024x4096) = simp[b] (1024x128, row-major) @ inc[b] (128x4096)
    gemm_wmma<128, 1, 0><<<dim3(64, 16, 4), gB, 0, stream>>>(
        simp, 131072, inc, 524288, scores, 1024L * 4096,
        1024, 128, nullptr, nullptr, nullptr, nullptr, 0);
    softmax_rows<<<4096, 256, 0, stream>>>(scores);
    // out[b] (128x4096) = simp[b]^T (128x1024, strided) @ scores[b] + inc[b]
    gemm_wmma<1, 128, EP_RESID><<<dim3(64, 2, 4), gB, 0, stream>>>(
        simp, 131072, scores, 1024L * 4096,
        acat + (br ? 524288L : 0), 2L * 524288,
        128, 1024, nullptr, nullptr, nullptr, inc, 524288);
  }

  // fused = stdconv1x1(acat) + dif
  gemm_wmma<256, 1, EP_STD | EP_RESID><<<dim3(64, 2, 4), gB, 0, stream>>>(
      fu_w, 0, acat, 2L * 524288, fusedb, 524288,
      128, 256, fu_b, fu_s, fu_o, difbuf, 524288);

  // bilinear upsample 64 -> 128
  upsample_bilinear<<<32768, 256, 0, stream>>>(fusedb, (float*)d_out);
}
